// MixtureOfExperts_42082089566762
// MI455X (gfx1250) — matrix operand order repaired and checked by reading, not compiled, not run on hardware
//
#include <hip/hip_runtime.h>

// Problem sizes (B=4, S=2048 -> 8192 tokens)
#define NTOK 8192
#define DD   1024
#define FF   4096
#define EE   8

typedef __attribute__((ext_vector_type(16))) __bf16         v16bf;
typedef __attribute__((ext_vector_type(8)))  float          v8f;
typedef __attribute__((ext_vector_type(4)))  unsigned int   uiv4;
typedef __attribute__((ext_vector_type(4)))  float          fv4;
typedef __attribute__((ext_vector_type(4)))  unsigned short usv4;
typedef int gv4 __attribute__((vector_size(16)));   // matches builtin's V4i

#define AS1 __attribute__((address_space(1)))
#define AS3 __attribute__((address_space(3)))

#if defined(__has_builtin)
#  if __has_builtin(__builtin_amdgcn_global_load_async_to_lds_b128)
#    define HAVE_ASYNC_LDS 1
#  endif
#endif

union FragBF { v16bf v; uiv4 q[2]; };

__device__ __forceinline__ unsigned short f2bf(float f) {
  unsigned int u = __float_as_uint(f);
  u += 0x7FFFu + ((u >> 16) & 1u);   // round-to-nearest-even
  return (unsigned short)(u >> 16);
}

// ---------------------------------------------------------------------------
// fp32 -> bf16 weight conversion (one-time pass; 4-wide vectorized)
// ---------------------------------------------------------------------------
__global__ void f32_to_bf16_kernel(const float* __restrict__ src,
                                   unsigned short* __restrict__ dst,
                                   long long n4) {
  long long i = (long long)blockIdx.x * blockDim.x + threadIdx.x;
  long long stride = (long long)gridDim.x * blockDim.x;
  for (; i < n4; i += stride) {
    fv4 v = reinterpret_cast<const fv4*>(src)[i];
    usv4 o;
    o.x = f2bf(v.x); o.y = f2bf(v.y); o.z = f2bf(v.z); o.w = f2bf(v.w);
    reinterpret_cast<usv4*>(dst)[i] = o;
  }
}

// ---------------------------------------------------------------------------
// Router: logits -> softmax -> top-2 -> renorm; append to per-expert lists.
// One wave32 per token. Also emits x in bf16 for the WMMA path.
// ---------------------------------------------------------------------------
__global__ __launch_bounds__(256)
void moe_router(const float* __restrict__ x,
                const float* __restrict__ Wr,
                unsigned short* __restrict__ xb,
                int* __restrict__ tlist,
                float* __restrict__ wlist,
                int* __restrict__ counts) {
  __shared__ float sWr[EE * DD];
  int tid = threadIdx.x;
  for (int i = tid; i < EE * DD; i += 256) sWr[i] = Wr[i];
  __syncthreads();

  int w = tid >> 5, l = tid & 31;
  int n = blockIdx.x * 8 + w;
  const float* xr = x + (size_t)n * DD;
  unsigned short* xbr = xb + (size_t)n * DD;

  float acc[EE];
#pragma unroll
  for (int e = 0; e < EE; ++e) acc[e] = 0.f;
  for (int d = l; d < DD; d += 32) {
    float xv = xr[d];
    xbr[d] = f2bf(xv);
#pragma unroll
    for (int e = 0; e < EE; ++e) acc[e] += xv * sWr[e * DD + d];
  }
#pragma unroll
  for (int e = 0; e < EE; ++e) {
#pragma unroll
    for (int off = 16; off > 0; off >>= 1)
      acc[e] += __shfl_xor(acc[e], off, 32);
  }
  if (l == 0) {
    float m = acc[0];
#pragma unroll
    for (int e = 1; e < EE; ++e) m = fmaxf(m, acc[e]);
    float p[EE];
#pragma unroll
    for (int e = 0; e < EE; ++e) p[e] = __expf(acc[e] - m);
    int i0 = 0;
#pragma unroll
    for (int e = 1; e < EE; ++e) if (p[e] > p[i0]) i0 = e;
    int i1 = (i0 == 0) ? 1 : 0;
#pragma unroll
    for (int e = 0; e < EE; ++e) if (e != i0 && p[e] > p[i1]) i1 = e;
    float s = p[i0] + p[i1];
    float w0 = p[i0] / s, w1 = p[i1] / s;
    int pos0 = atomicAdd(&counts[i0], 1);
    tlist[i0 * NTOK + pos0] = n;             // slot 0
    wlist[i0 * NTOK + pos0] = w0;
    int pos1 = atomicAdd(&counts[i1], 1);
    tlist[i1 * NTOK + pos1] = n | (1 << 16); // slot 1
    wlist[i1 * NTOK + pos1] = w1;
  }
}

// ---------------------------------------------------------------------------
// Fused expert FFN: gate/up GEMM (K=1024) -> SiLU*up -> down GEMM (K=F),
// all on v_wmma_f32_16x16x32_bf16. Block = 256 thr (8 wave32), 32 tokens.
// The gathered x tile is staged into LDS once via async-to-LDS DMA.
// ---------------------------------------------------------------------------
__global__ __launch_bounds__(256)
void moe_ffn(const unsigned short* __restrict__ xb,
             const unsigned short* __restrict__ wg,
             const unsigned short* __restrict__ wu,
             const unsigned short* __restrict__ wd,
             const int* __restrict__ tlist,
             const float* __restrict__ wlist,
             const int* __restrict__ counts,
             float* __restrict__ yws) {
  int e = blockIdx.y;
  int cnt = counts[e];
  int tile0 = blockIdx.x * 32;
  if (tile0 >= cnt) return;              // uniform exit: EXEC stays all-ones

  __shared__ unsigned short x_lds[32][1032]; // gathered x tile (bf16, padded)
  __shared__ float hg[32][33];
  __shared__ float hu[32][33];
  __shared__ unsigned short hbf[32][40]; // h-slab, bf16, padded rows
  __shared__ int   sEnt[32];
  __shared__ float sWt[32];

  int tid = threadIdx.x;
  if (tid < 32) {
    int idx = tile0 + tid;
    int cidx = idx < cnt ? idx : cnt - 1;
    sEnt[tid] = tlist[(size_t)e * NTOK + cidx];
    sWt[tid]  = wlist[(size_t)e * NTOK + cidx];
  }
  __syncthreads();

  int w = tid >> 5, l = tid & 31;
  int h = l >> 4, ln = l & 15;
  int wtype = w >> 2;                    // 0 = gate, 1 = up
  int qm = w & 1, qf = (w >> 1) & 1;     // token-half, f-half quadrant

  // ---- stage gathered x rows into LDS (async DMA; ASYNCcnt-tracked) ----
  {
#pragma unroll
    for (int rr = 0; rr < 4; ++rr) {
      int r = w * 4 + rr;                          // wave owns 4 rows
      const unsigned short* grow = xb + (size_t)(sEnt[r] & 0xFFFF) * DD;
#pragma unroll
      for (int j = 0; j < 4; ++j) {
        int c8 = (j * 32 + l) * 8;                 // 16B chunk per lane
#ifdef HAVE_ASYNC_LDS
        __builtin_amdgcn_global_load_async_to_lds_b128(
            (AS1 gv4*)(size_t)(grow + c8),
            (AS3 gv4*)(unsigned int)(size_t)&x_lds[r][c8],
            0, 0);
#else
        *reinterpret_cast<uiv4*>(&x_lds[r][c8]) =
            *reinterpret_cast<const uiv4*>(grow + c8);
#endif
      }
    }
#ifdef HAVE_ASYNC_LDS
#  if __has_builtin(__builtin_amdgcn_s_wait_asynccnt)
    __builtin_amdgcn_s_wait_asynccnt(0);
#  else
    asm volatile("s_wait_asynccnt 0" ::: "memory");
#  endif
#endif
  }
  __syncthreads();

  const unsigned short* wsel = (wtype ? wu : wg) + (size_t)e * FF * DD;
  const unsigned short* wde  = wd + (size_t)e * (size_t)DD * FF;
  const unsigned short* xrow_lds = &x_lds[qm * 16 + ln][0];

  int dbase = w * 128;                   // this wave's D slice for down-proj

  v8f zero = {};
  v8f acc0[8], acc1[8];
#pragma unroll
  for (int c = 0; c < 8; ++c) { acc0[c] = zero; acc1[c] = zero; }

  for (int fb = 0; fb < FF; fb += 32) {
    // ---- gate/up: 16x16 tile, K = D = 1024 (32 WMMAs), A from LDS ----
    v8f g = zero;
    const unsigned short* brow = wsel + (size_t)(fb + qf * 16 + ln) * DD;
#pragma unroll 4
    for (int kk = 0; kk < DD; kk += 32) {
      FragBF a, b;
      // A (16x32 bf16): lane=M row, 16 contiguous K per half
      a.q[0] = *reinterpret_cast<const uiv4*>(xrow_lds + kk + h * 16);
      a.q[1] = *reinterpret_cast<const uiv4*>(xrow_lds + kk + h * 16 + 8);
      // B (32x16 bf16): lane=N column, 16 contiguous K per half
      const uiv4* bp = reinterpret_cast<const uiv4*>(brow + kk + h * 16);
      b.q[0] = bp[0];
      b.q[1] = bp[1];
      g = __builtin_amdgcn_wmma_f32_16x16x32_bf16(false, a.v, false, b.v,
                                                  (short)0, g, false, false);
    }
    __builtin_prefetch(brow + 32 * DD, 0, 1);  // next F-slab weight row

    // stage gate/up tile to LDS
    {
      float (*dst)[33] = wtype ? hu : hg;
#pragma unroll
      for (int i = 0; i < 8; ++i)
        dst[qm * 16 + i + h * 8][qf * 16 + ln] = g[i];
    }
    __syncthreads();

    // ---- h = silu(gate) * up -> bf16 slab (fast rcp path) ----
#pragma unroll
    for (int j = 0; j < 4; ++j) {
      int idx = tid + j * 256;
      int r = idx >> 5, c = idx & 31;
      float gv = hg[r][c], uv = hu[r][c];
      float sv = gv * __builtin_amdgcn_rcpf(1.f + __expf(-gv));
      hbf[r][c] = f2bf(sv * uv);
    }
    __syncthreads();

    // ---- down-proj: accumulate [32 tok x 128 d] slice, K = 32 ----
    FragBF a0, a1;
    a0.q[0] = *reinterpret_cast<const uiv4*>(&hbf[ln][h * 16]);
    a0.q[1] = *reinterpret_cast<const uiv4*>(&hbf[ln][h * 16 + 8]);
    a1.q[0] = *reinterpret_cast<const uiv4*>(&hbf[16 + ln][h * 16]);
    a1.q[1] = *reinterpret_cast<const uiv4*>(&hbf[16 + ln][h * 16 + 8]);
#pragma unroll
    for (int c = 0; c < 8; ++c) {
      FragBF b;
      const uiv4* bp = reinterpret_cast<const uiv4*>(
          wde + (size_t)(dbase + c * 16 + ln) * FF + fb + h * 16);
      b.q[0] = bp[0];
      b.q[1] = bp[1];
      acc0[c] = __builtin_amdgcn_wmma_f32_16x16x32_bf16(false, a0.v, false, b.v,
                                                        (short)0, acc0[c], false, false);
      acc1[c] = __builtin_amdgcn_wmma_f32_16x16x32_bf16(false, a1.v, false, b.v,
                                                        (short)0, acc1[c], false, false);
    }
    __syncthreads();
  }

  // ---- store: weighted scatter to per-(token,slot) rows (no atomics) ----
#pragma unroll
  for (int mh = 0; mh < 2; ++mh) {
#pragma unroll
    for (int i = 0; i < 8; ++i) {
      int rr = mh * 16 + i + h * 8;                 // row within 32-token tile
      if (tile0 + rr < cnt) {
        int ent2 = sEnt[rr];
        float wt = sWt[rr];
        int t2 = ent2 & 0xFFFF;
        int slot = (ent2 >> 16) & 1;
        float* yrow = yws + ((size_t)t2 * 2 + slot) * DD + dbase;
#pragma unroll
        for (int c = 0; c < 8; ++c)
          yrow[c * 16 + ln] = (mh ? acc1[c][i] : acc0[c][i]) * wt;
      }
    }
  }
}

// ---------------------------------------------------------------------------
// Combine: out = y[slot0] + y[slot1]
// ---------------------------------------------------------------------------
__global__ void moe_combine(const float* __restrict__ yws,
                            float* __restrict__ out) {
  long long i = (long long)blockIdx.x * blockDim.x + threadIdx.x;
  long long total = (long long)NTOK * DD / 4;
  if (i >= total) return;
  long long idx = i * 4;
  long long n = idx >> 10;
  long long d = idx & (DD - 1);
  fv4 y0 = *reinterpret_cast<const fv4*>(yws + (n * 2) * DD + d);
  fv4 y1 = *reinterpret_cast<const fv4*>(yws + (n * 2 + 1) * DD + d);
  *reinterpret_cast<fv4*>(out + n * DD + d) = y0 + y1;
}

// ---------------------------------------------------------------------------
extern "C" void kernel_launch(void* const* d_in, const int* in_sizes, int n_in,
                              void* d_out, int out_size, void* d_ws, size_t ws_size,
                              hipStream_t stream) {
  (void)in_sizes; (void)n_in; (void)out_size; (void)ws_size;
  const float* x  = (const float*)d_in[0];
  const float* Wr = (const float*)d_in[1];
  const float* Wg = (const float*)d_in[2];
  const float* Wu = (const float*)d_in[3];
  const float* Wd = (const float*)d_in[4];
  float* out = (float*)d_out;

  char* ws = (char*)d_ws;
  size_t o = 0;
  unsigned short* xb  = (unsigned short*)(ws + o); o += (size_t)NTOK * DD * 2;
  unsigned short* wgb = (unsigned short*)(ws + o); o += (size_t)EE * FF * DD * 2;
  unsigned short* wub = (unsigned short*)(ws + o); o += (size_t)EE * FF * DD * 2;
  unsigned short* wdb = (unsigned short*)(ws + o); o += (size_t)EE * DD * FF * 2;
  float* yws   = (float*)(ws + o); o += (size_t)NTOK * 2 * DD * 4;
  int*   tlist = (int*)  (ws + o); o += (size_t)EE * NTOK * 4;
  float* wlist = (float*)(ws + o); o += (size_t)EE * NTOK * 4;
  int*  counts = (int*)  (ws + o); o += 256;

  (void)hipMemsetAsync(counts, 0, EE * sizeof(int), stream);

  long long wq = (long long)EE * FF * DD / 4;
  f32_to_bf16_kernel<<<4096, 256, 0, stream>>>(Wg, wgb, wq);
  f32_to_bf16_kernel<<<4096, 256, 0, stream>>>(Wu, wub, wq);
  f32_to_bf16_kernel<<<4096, 256, 0, stream>>>(Wd, wdb, wq);

  moe_router<<<NTOK / 8, 256, 0, stream>>>(x, Wr, xb, tlist, wlist, counts);

  moe_ffn<<<dim3(NTOK / 32, EE), 256, 0, stream>>>(xb, wgb, wub, wdb,
                                                   tlist, wlist, counts, yws);

  moe_combine<<<((long long)NTOK * DD / 4 + 255) / 256, 256, 0, stream>>>(yws, out);
}
